// MultiResBiplane_79164837200479
// MI455X (gfx1250) — matrix operand order repaired and checked by reading, not compiled, not run on hardware
//
#include <hip/hip_runtime.h>
#include <hip/hip_bf16.h>

typedef __attribute__((ext_vector_type(16))) _Float16 v16h;
typedef __attribute__((ext_vector_type(8)))  _Float16 v8h;
typedef __attribute__((ext_vector_type(8)))  float    v8f;

#define P_TOT   (8*512*512)
#define IN_DIM  88
#define K_PAD   96
#define HID     44
#define N_PAD   48
#define WAVES_PER_BLK 8

// Transposed-table layout (halfs): level l plane stored [R][R][16] f16
// offsets: {0, 64^2*16, +128^2*16, +256^2*16} = {0, 65536, 327680, 1376256}
__global__ __launch_bounds__(256)
void transpose_embs_kernel(const float* __restrict__ e0, const float* __restrict__ e1,
                           const float* __restrict__ e2, const float* __restrict__ e3,
                           _Float16* __restrict__ T) {
    int idx = blockIdx.x * blockDim.x + threadIdx.x;
    const float* src; int R2; long dstOff; int loc;
    if      (idx < 4096)   { src = e0; loc = idx;         R2 = 4096;   dstOff = 0;       }
    else if (idx < 20480)  { src = e1; loc = idx - 4096;  R2 = 16384;  dstOff = 65536;   }
    else if (idx < 86016)  { src = e2; loc = idx - 20480; R2 = 65536;  dstOff = 327680;  }
    else if (idx < 348160) { src = e3; loc = idx - 86016; R2 = 262144; dstOff = 1376256; }
    else return;
    v8h lo, hi;
#pragma unroll
    for (int c = 0; c < 8; ++c) {
        lo[c] = (_Float16)src[(long)c * R2 + loc];
        hi[c] = (_Float16)src[(long)(c + 8) * R2 + loc];
    }
    v8h* dst = (v8h*)(T + dstOff + ((long)loc << 4));
    dst[0] = lo;
    dst[1] = hi;
}

__global__ __launch_bounds__(256)
void biplane_mlp_kernel(const float* __restrict__ coords, const float* __restrict__ tfeat,
                        const float* __restrict__ W1, const float* __restrict__ b1,
                        const float* __restrict__ W2, const float* __restrict__ b2,
                        const _Float16* __restrict__ T, float* __restrict__ out,
                        int numTiles) {
    __shared__ float    sW2[HID * 2];
    __shared__ float    sB1[N_PAD];
    __shared__ float    sB2[2];
    __shared__ _Float16 sX[WAVES_PER_BLK][16 * K_PAD];   // 24 KB
    __shared__ float    sH[WAVES_PER_BLK][16 * N_PAD];   // 24 KB

    const int tid = threadIdx.x;
    if (tid < HID * 2) sW2[tid] = W2[tid];
    if (tid < N_PAD)   sB1[tid] = (tid < HID) ? b1[tid] : 0.0f;
    if (tid < 2)       sB2[tid] = b2[tid];
    __syncthreads();

    const int  wid  = tid >> 5;
    const int  lane = tid & 31;
    const int  l15  = lane & 15;
    const bool hiH  = lane >= 16;

    // ---- Preload W1 as f16 B-matrix fragments (3 K-tiles x 3 N-tiles), once per wave.
    // B 32x16: lanes 0-15 hold K={0..7,16..23} (col N=lane), lanes 16-31 hold K={8..15,24..31} (col N=lane-16).
    v16h Bfrag[3][3];
#pragma unroll
    for (int kt = 0; kt < 3; ++kt)
#pragma unroll
        for (int nt = 0; nt < 3; ++nt) {
#pragma unroll
            for (int i = 0; i < 16; ++i) {
                int K = kt * 32 + (hiH ? 8 : 0) + (i < 8 ? i : i + 8);
                int N = nt * 16 + l15;
                float w = (K < IN_DIM && N < HID) ? W1[K * HID + N] : 0.0f;
                Bfrag[kt][nt][i] = (_Float16)w;
            }
        }

    _Float16* X = sX[wid];
    float*    H = sH[wid];

    const int Rtab[4] = {64, 128, 256, 512};
    const long Otab[4] = {0, 65536, 327680, 1376256};

    const int totalWaves = gridDim.x * WAVES_PER_BLK;
    for (int tile = blockIdx.x * WAVES_PER_BLK + wid; tile < numTiles; tile += totalWaves) {
        const int base = tile << 4;            // 16 points per tile
        const int bat  = base >> 18;           // 512*512 = 2^18 points per batch (tile-uniform)
        const int m    = l15;                  // point-in-tile handled by this lane
        const int p    = base + m;
        const float cx = coords[2 * p + 0];
        const float cy = coords[2 * p + 1];

        // ---- grid-sample two levels per lane-half into LDS X tile (f16) ----
#pragma unroll
        for (int s = 0; s < 2; ++s) {
            const int  lvl = (hiH ? 2 : 0) + s;
            const int  R   = Rtab[lvl];
            const long off = Otab[lvl];

            float fx = (cx + 1.0f) * 0.5f * (float)(R - 1);
            float fy = (cy + 1.0f) * 0.5f * (float)(R - 1);
            float x0f = floorf(fx), y0f = floorf(fy);
            float wx1 = fx - x0f, wy1 = fy - y0f;
            float wx0 = 1.0f - wx1, wy0 = 1.0f - wy1;
            int x0 = (int)x0f, y0 = (int)y0f;
            int x1 = x0 + 1,  y1 = y0 + 1;
            float vx0 = (x0 >= 0 && x0 < R) ? 1.0f : 0.0f;
            float vx1 = (x1 >= 0 && x1 < R) ? 1.0f : 0.0f;
            float vy0 = (y0 >= 0 && y0 < R) ? 1.0f : 0.0f;
            float vy1 = (y1 >= 0 && y1 < R) ? 1.0f : 0.0f;
            int cx0 = min(max(x0, 0), R - 1), cx1 = min(max(x1, 0), R - 1);
            int cy0 = min(max(y0, 0), R - 1), cy1 = min(max(y1, 0), R - 1);
            float w00 = wy0 * wx0 * vy0 * vx0, w01 = wy0 * wx1 * vy0 * vx1;
            float w10 = wy1 * wx0 * vy1 * vx0, w11 = wy1 * wx1 * vy1 * vx1;

            const _Float16* t00 = T + off + ((long)(cy0 * R + cx0) << 4);
            const _Float16* t01 = T + off + ((long)(cy0 * R + cx1) << 4);
            const _Float16* t10 = T + off + ((long)(cy1 * R + cx0) << 4);
            const _Float16* t11 = T + off + ((long)(cy1 * R + cx1) << 4);
            v8h a00 = *(const v8h*)t00,       a01 = *(const v8h*)t01;
            v8h a10 = *(const v8h*)t10,       a11 = *(const v8h*)t11;
            v8h b00 = *(const v8h*)(t00 + 8), b01 = *(const v8h*)(t01 + 8);
            v8h b10 = *(const v8h*)(t10 + 8), b11 = *(const v8h*)(t11 + 8);

            _Float16* xrow = X + m * K_PAD + lvl * 16;
#pragma unroll
            for (int c = 0; c < 8; ++c) {
                float vlo = w00 * (float)a00[c] + w01 * (float)a01[c]
                          + w10 * (float)a10[c] + w11 * (float)a11[c];
                float vhi = w00 * (float)b00[c] + w01 * (float)b01[c]
                          + w10 * (float)b10[c] + w11 * (float)b11[c];
                xrow[c]     = (_Float16)vlo;
                xrow[c + 8] = (_Float16)vhi;
            }
        }
        if (hiH) {
            _Float16* xrow = X + m * K_PAD + 64;
#pragma unroll
            for (int j = 0; j < 24; ++j) xrow[j] = (_Float16)tfeat[bat * 24 + j];
#pragma unroll
            for (int j = 24; j < 32; ++j) xrow[j] = (_Float16)0.0f;
        }
        // intra-wave producer->consumer on LDS (split dep counters on gfx1250)
        asm volatile("s_wait_dscnt 0" ::: "memory");

        // ---- A fragments: 16-bit A 16x32 layout.
        // lanes 0-15: M=lane, K = kt*32 + {0..7, 16..23}; lanes 16-31: M=lane-16, K += 8.
        v16h A[3];
#pragma unroll
        for (int kt = 0; kt < 3; ++kt) {
            const _Float16* row = X + m * K_PAD + kt * 32 + (hiH ? 8 : 0);
#pragma unroll
            for (int i = 0; i < 8; ++i) {
                A[kt][i]     = row[i];
                A[kt][i + 8] = row[16 + i];
            }
        }

        // ---- Layer 1 on the matrix pipe: 3 N-tiles x 3 K-tiles of v_wmma_f32_16x16x32_f16
#pragma unroll
        for (int nt = 0; nt < 3; ++nt) {
            v8f acc = {};
#pragma unroll
            for (int kt = 0; kt < 3; ++kt)
                acc = __builtin_amdgcn_wmma_f32_16x16x32_f16(
                    false, A[kt], false, Bfrag[kt][nt], (short)0, acc, false, false);
            const int N = nt * 16 + l15;
            const float bb = sB1[N];
#pragma unroll
            for (int r = 0; r < 8; ++r) {            // C/D layout: VGPR r -> row M=r (+8 for hi lanes)
                const int M = r + (hiH ? 8 : 0);
                float h = acc[r] + bb;
                h = (h > 0.0f) ? h : 0.01f * h;       // LeakyReLU
                H[M * N_PAD + N] = h;
            }
        }
        asm volatile("s_wait_dscnt 0" ::: "memory");

        // ---- Layer 2 + sigmoid: 32 lanes = 16 points x 2 outputs
        const int m2 = lane >> 1;
        const int o  = lane & 1;
        float a2 = sB2[o];
#pragma unroll
        for (int k = 0; k < HID; ++k) a2 += H[m2 * N_PAD + k] * sW2[k * 2 + o];
        out[(long)(base + m2) * 2 + o] = 1.0f / (1.0f + __expf(-a2));
    }
}

extern "C" void kernel_launch(void* const* d_in, const int* in_sizes, int n_in,
                              void* d_out, int out_size, void* d_ws, size_t ws_size,
                              hipStream_t stream) {
    const float* coords = (const float*)d_in[0];
    const float* tfeat  = (const float*)d_in[1];
    const float* e0     = (const float*)d_in[2];
    const float* e1     = (const float*)d_in[3];
    const float* e2     = (const float*)d_in[4];
    const float* e3     = (const float*)d_in[5];
    const float* W1     = (const float*)d_in[6];
    const float* b1     = (const float*)d_in[7];
    const float* W2     = (const float*)d_in[8];
    const float* b2     = (const float*)d_in[9];
    _Float16* T  = (_Float16*)d_ws;   // ~10.6 MB f16 transposed tables
    float*    out = (float*)d_out;

    transpose_embs_kernel<<<(348160 + 255) / 256, 256, 0, stream>>>(e0, e1, e2, e3, T);

    const int numTiles = P_TOT / 16;  // 131072 wave-tiles
    biplane_mlp_kernel<<<2048, 256, 0, stream>>>(coords, tfeat, W1, b1, W2, b2, T, out, numTiles);
}
